// CE_15238543966409
// MI455X (gfx1250) — matrix operand order repaired and checked by reading, not compiled, not run on hardware
//
#include <hip/hip_runtime.h>
#include <hip/hip_bf16.h>

// ---------------------------------------------------------------------------
// MI455X (gfx1250) patch-attention block, bf16 WMMA pipeline + async-LDS DMA.
// Shapes: B=2, Cin=64, H=W=256, Ci=16, K=7, S=4, Lh=Lw=64, L=4096,
//         d_in=784 (pad 800), d_out=196 (pad 224). pt=pl=1, pb=pr=2.
// ---------------------------------------------------------------------------

typedef unsigned short u16;
typedef u16   v8u16  __attribute__((ext_vector_type(8)));
typedef u16   v16u16 __attribute__((ext_vector_type(16)));
typedef __bf16 bf16;
typedef bf16  v16bf  __attribute__((ext_vector_type(16)));
typedef float v8f    __attribute__((ext_vector_type(8)));

__device__ __forceinline__ u16 f2bf(float f) {
  unsigned u = __float_as_uint(f);
  u += 0x7FFFu + ((u >> 16) & 1u);          // round-to-nearest-even
  return (u16)(u >> 16);
}

// A-operand fragment (16x32 bf16, MxK): lane<16 -> M=lane, K {0..7}U{16..23};
// lane>=16 -> M=lane&15, K {8..15}U{24..31}. Storage row-major [m][ld].
__device__ __forceinline__ v16bf frag_a_ld(const u16* tile, int ld, int kb, int lane) {
  int hi = (lane >> 4) & 1;
  int r  = lane & 15;
  const u16* q = tile + (size_t)r * ld + kb + hi * 8;
  v8u16 c0 = *(const v8u16*)q;
  v8u16 c1 = *(const v8u16*)(q + 16);
  v16u16 o;
#pragma unroll
  for (int i = 0; i < 8; i++) { o[i] = c0[i]; o[i + 8] = c1[i]; }
  return __builtin_bit_cast(v16bf, o);
}

// B-operand fragment (32x16 bf16, KxN): lane holds column n=lane&15,
// K {8hi..8hi+7}U{16+8hi..16+8hi+7}. Storage row-major [n][ld] (i.e. B^T).
__device__ __forceinline__ v16bf frag_b_ld(const u16* tile, int ld, int kb, int lane) {
  int hi = (lane >> 4) & 1;
  int n  = lane & 15;
  const u16* q = tile + (size_t)n * ld + kb + hi * 8;
  v8u16 c0 = *(const v8u16*)q;
  v8u16 c1 = *(const v8u16*)(q + 16);
  v16u16 o;
#pragma unroll
  for (int i = 0; i < 8; i++) { o[i] = c0[i]; o[i + 8] = c1[i]; }
  return __builtin_bit_cast(v16bf, o);
}

__device__ __forceinline__ v8f wmma_bf16(v16bf a, v16bf b, v8f c) {
  return __builtin_amdgcn_wmma_f32_16x16x32_bf16(false, a, false, b, (short)0, c,
                                                 false, false);
}

// Async global->LDS 16-byte copy (CDNA5 vflat async op, tracked by ASYNCcnt).
// lds_off is the byte offset within the workgroup's LDS allocation.
__device__ __forceinline__ void async_cp16(unsigned lds_off, const void* gsrc) {
  asm volatile("global_load_async_to_lds_b128 %0, %1, off"
               :: "v"(lds_off), "v"(gsrc)
               : "memory");
}
__device__ __forceinline__ void wait_async_le2() {
  asm volatile("s_wait_asynccnt 0x2" ::: "memory");
}
__device__ __forceinline__ void wait_async_0() {
  asm volatile("s_wait_asynccnt 0x0" ::: "memory");
}

// reductions across the 16 lanes of each wave-half (C/D layout halves)
__device__ __forceinline__ float redmax16(float v) {
  v = fmaxf(v, __shfl_xor(v, 1));
  v = fmaxf(v, __shfl_xor(v, 2));
  v = fmaxf(v, __shfl_xor(v, 4));
  v = fmaxf(v, __shfl_xor(v, 8));
  return v;
}
__device__ __forceinline__ float redsum16(float v) {
  v += __shfl_xor(v, 1);
  v += __shfl_xor(v, 2);
  v += __shfl_xor(v, 4);
  v += __shfl_xor(v, 8);
  return v;
}

// ---------------------------------------------------------------------------
// 1) b1 = conv3x3(b, g_w)+g_b ; b2 = conv1x1(b, theta_w)+theta_b  -> bf16
// ---------------------------------------------------------------------------
__global__ void k_conv(const float* __restrict__ bin, const float* __restrict__ gw_g,
                       const float* __restrict__ gb_g, const float* __restrict__ tw_g,
                       const float* __restrict__ tb_g, u16* __restrict__ b1o,
                       u16* __restrict__ b2o) {
  __shared__ float gw[16 * 64 * 9];
  __shared__ float tw[16 * 64];
  __shared__ float gb[16], tb[16];
  for (int i = threadIdx.x; i < 16 * 64 * 9; i += blockDim.x) gw[i] = gw_g[i];
  for (int i = threadIdx.x; i < 16 * 64; i += blockDim.x) tw[i] = tw_g[i];
  if (threadIdx.x < 16) { gb[threadIdx.x] = gb_g[threadIdx.x]; tb[threadIdx.x] = tb_g[threadIdx.x]; }
  __syncthreads();

  int pix = blockIdx.x * blockDim.x + threadIdx.x;   // B*H*W = 131072
  int b = pix >> 16, y = (pix >> 8) & 255, x = pix & 255;
  float a1[16], a2[16];
#pragma unroll
  for (int co = 0; co < 16; co++) { a1[co] = gb[co]; a2[co] = tb[co]; }
  const float* bp = bin + (size_t)b * 64 * 65536;
  for (int ci = 0; ci < 64; ci++) {
    const float* pl = bp + (size_t)ci * 65536;
    float v[9];
#pragma unroll
    for (int ky = 0; ky < 3; ky++)
#pragma unroll
      for (int kx = 0; kx < 3; kx++) {
        int yy = y + ky - 1, xx = x + kx - 1;
        v[ky * 3 + kx] = ((unsigned)yy < 256u && (unsigned)xx < 256u)
                             ? pl[yy * 256 + xx] : 0.f;
      }
    float c = v[4];
#pragma unroll
    for (int co = 0; co < 16; co++) {
      const float* w = &gw[(co * 64 + ci) * 9];
      float s = a1[co];
#pragma unroll
      for (int kk = 0; kk < 9; kk++) s = fmaf(w[kk], v[kk], s);
      a1[co] = s;
      a2[co] = fmaf(tw[co * 64 + ci], c, a2[co]);
    }
  }
#pragma unroll
  for (int co = 0; co < 16; co++) {
    size_t o = ((size_t)(b * 16 + co) * 256 + y) * 256 + x;
    b1o[o] = f2bf(a1[co]);
    b2o[o] = f2bf(a2[co]);
  }
}

// ---------------------------------------------------------------------------
// 2) thr/bias: 7x7 stride-4 convs on padded b -> (B,4096) each
// ---------------------------------------------------------------------------
__global__ void k_thrbias(const float* __restrict__ bin, const float* __restrict__ tw_g,
                          const float* __restrict__ tb_g, const float* __restrict__ bw_g,
                          const float* __restrict__ bb_g, float* __restrict__ thr,
                          float* __restrict__ bias) {
  __shared__ float tw[64 * 49], bw[64 * 49];
  for (int i = threadIdx.x; i < 64 * 49; i += blockDim.x) { tw[i] = tw_g[i]; bw[i] = bw_g[i]; }
  __syncthreads();
  int idx = blockIdx.x * blockDim.x + threadIdx.x;   // 8192
  int b = idx >> 12, li = idx & 4095, lh = li >> 6, lw = li & 63;
  float at = tb_g[0], ab = bb_g[0];
  const float* bp = bin + (size_t)b * 64 * 65536;
  for (int ci = 0; ci < 64; ci++) {
    const float* pl = bp + (size_t)ci * 65536;
#pragma unroll
    for (int ky = 0; ky < 7; ky++) {
      int yy = lh * 4 + ky - 1;
      if ((unsigned)yy >= 256u) continue;
#pragma unroll
      for (int kx = 0; kx < 7; kx++) {
        int xx = lw * 4 + kx - 1;
        if ((unsigned)xx >= 256u) continue;
        float v = pl[yy * 256 + xx];
        at = fmaf(tw[ci * 49 + ky * 7 + kx], v, at);
        ab = fmaf(bw[ci * 49 + ky * 7 + kx], v, ab);
      }
    }
  }
  thr[idx] = at;
  bias[idx] = ab;
}

// ---------------------------------------------------------------------------
// 3) pad/convert fc weights -> bf16 [2][224][800], bias -> f32 [2][224]
// ---------------------------------------------------------------------------
__global__ void k_prep(const float* __restrict__ fc1w, const float* __restrict__ fc1b,
                       const float* __restrict__ fc2w, const float* __restrict__ fc2b,
                       u16* __restrict__ fcw, float* __restrict__ fcb) {
  int idx = blockIdx.x * blockDim.x + threadIdx.x;   // 2*224*800 = 358400
  int f = idx / (224 * 800);
  int rem = idx - f * 224 * 800;
  int o = rem / 800, k = rem - o * 800;
  const float* src = f ? fc2w : fc1w;
  float v = (o < 196 && k < 784) ? src[o * 784 + k] : 0.f;
  fcw[idx] = f2bf(v);
  if (idx < 2 * 224) {
    int ff = idx / 224, oo = idx - ff * 224;
    const float* sb = ff ? fc2b : fc1b;
    fcb[idx] = (oo < 196) ? sb[oo] : 0.f;
  }
}

// ---------------------------------------------------------------------------
// 4) pi_t[b][d][l] = patch(b2)  (pre-transposed for agg B-operand)
// ---------------------------------------------------------------------------
__global__ void k_patch_pi(const u16* __restrict__ b2, u16* __restrict__ pit) {
  size_t idx = (size_t)blockIdx.x * blockDim.x + threadIdx.x;   // 2*784*4096
  int b = (int)(idx / (784u * 4096u));
  int rem = (int)(idx - (size_t)b * 784 * 4096);
  int d = rem >> 12, l = rem & 4095;
  int ci = d / 49, r2 = d - ci * 49, ky = r2 / 7, kx = r2 - ky * 7;
  int lh = l >> 6, lw = l & 63;
  int y = lh * 4 + ky - 1, x = lw * 4 + kx - 1;
  u16 v = 0;
  if ((unsigned)y < 256u && (unsigned)x < 256u)
    v = b2[((size_t)(b * 16 + ci) * 256 + y) * 256 + x];
  pit[idx] = v;
}

// ---------------------------------------------------------------------------
// 5) fc GEMM: wi_f = relu(patch(b1)@fc1^T+b1f), xi_f likewise (shared A tile)
//    block = 7 waves; LDS holds one 16x800 bf16 patch tile
// ---------------------------------------------------------------------------
__global__ void __launch_bounds__(224)
k_fc(const u16* __restrict__ b1, const u16* __restrict__ fcw,
     const float* __restrict__ fcb, u16* __restrict__ wif, u16* __restrict__ xif) {
  __shared__ __align__(16) u16 A[16 * 800];
  int tile = blockIdx.x;   // 0..511 (16 rows of B*L each)
  for (int i = threadIdx.x; i < 16 * 800; i += blockDim.x) {
    int row = i / 800, k = i - row * 800;
    u16 v = 0;
    if (k < 784) {
      int ci = k / 49, r2 = k - ci * 49, ky = r2 / 7, kx = r2 - ky * 7;
      int l = tile * 16 + row, b = l >> 12, li = l & 4095;
      int lh = li >> 6, lw = li & 63;
      int y = lh * 4 + ky - 1, x = lw * 4 + kx - 1;
      if ((unsigned)y < 256u && (unsigned)x < 256u)
        v = b1[((size_t)(b * 16 + ci) * 256 + y) * 256 + x];
    }
    A[i] = v;
  }
  __syncthreads();

  int lane = threadIdx.x & 31, w = threadIdx.x >> 5, hi = lane >> 4;
  v8f acc[2][2] = {{{}, {}}, {{}, {}}};
  for (int kk = 0; kk < 25; kk++) {
    int kb = kk * 32;
    v16bf a = frag_a_ld(A, 800, kb, lane);
#pragma unroll
    for (int t = 0; t < 2; t++) {
      int n0 = (2 * w + t) * 16;
#pragma unroll
      for (int f = 0; f < 2; f++) {
        v16bf bf = frag_b_ld(fcw + (size_t)(f * 224 + n0) * 800, 800, kb, lane);
        acc[f][t] = wmma_bf16(a, bf, acc[f][t]);
      }
    }
  }
#pragma unroll
  for (int t = 0; t < 2; t++) {
    int col = (2 * w + t) * 16 + (lane & 15);
#pragma unroll
    for (int f = 0; f < 2; f++) {
      u16* out = f ? xif : wif;
      float bsum = fcb[f * 224 + col];
#pragma unroll
      for (int r = 0; r < 8; r++) {
        int grow = tile * 16 + r + 8 * hi;
        float vv = acc[f][t][r] + bsum;
        vv = vv > 0.f ? vv : 0.f;
        out[(size_t)grow * 224 + col] = f2bf(vv);
      }
    }
  }
}

// ---------------------------------------------------------------------------
// 6) attention: score -> mask/thr/bias -> masked softmax + renorm -> attn bf16
//    one wave owns 16 query rows; pass1 = online stats, pass2 = write attn
// ---------------------------------------------------------------------------
__global__ void k_attn(const u16* __restrict__ wif, const u16* __restrict__ xif,
                       const float* __restrict__ thr, const float* __restrict__ bias,
                       u16* __restrict__ attn) {
  int gw_ = blockIdx.x * (blockDim.x >> 5) + (threadIdx.x >> 5);   // 0..511
  int lane = threadIdx.x & 31, hi = lane >> 4;
  int b = gw_ >> 8, lbase = (gw_ & 255) * 16;

  const u16* arow = wif + (size_t)(b * 4096 + lbase) * 224;
  v16bf afr[7];
#pragma unroll
  for (int kk = 0; kk < 7; kk++) afr[kk] = frag_a_ld(arow, 224, kk * 32, lane);

  float thr_r[8], bias_r[8];
#pragma unroll
  for (int r = 0; r < 8; r++) {
    int l = b * 4096 + lbase + hi * 8 + r;
    thr_r[r] = thr[l];
    bias_r[r] = bias[l];
  }
  float rmx[8], sa[8], sm[8];
#pragma unroll
  for (int r = 0; r < 8; r++) { rmx[r] = -3.0e38f; sa[r] = 0.f; sm[r] = 0.f; }

  const u16* xbase = xif + (size_t)b * 4096 * 224;
  for (int mt = 0; mt < 256; mt++) {
    v8f acc = {};
    const u16* xb = xbase + (size_t)mt * 16 * 224;
#pragma unroll
    for (int kk = 0; kk < 7; kk++)
      acc = wmma_bf16(afr[kk], frag_b_ld(xb, 224, kk * 32, lane), acc);
#pragma unroll
    for (int r = 0; r < 8; r++) {
      float s = acc[r];
      bool mk = s >= thr_r[r];
      float g = (mk ? s : 0.f) + bias_r[r];
      float a = g * 10.0f;                        // SCALE
      float tm = redmax16(a);
      float nm = fmaxf(rmx[r], tm);
      float sc = __expf(rmx[r] - nm);
      float e = __expf(a - nm);
      float eAll = redsum16(e);
      float eM = redsum16(mk ? e : 0.f);
      sa[r] = sa[r] * sc + eAll;
      sm[r] = sm[r] * sc + eM;
      rmx[r] = nm;
    }
  }
  // attn[m] = exp(a-M)*mask / (sumMask + 1e-8*sumAll)
  float dinv[8];
#pragma unroll
  for (int r = 0; r < 8; r++) dinv[r] = 1.0f / (sm[r] + 1e-8f * sa[r]);

  u16* aout = attn + (size_t)(b * 4096 + lbase) * 4096;
  for (int mt = 0; mt < 256; mt++) {
    v8f acc = {};
    const u16* xb = xbase + (size_t)mt * 16 * 224;
#pragma unroll
    for (int kk = 0; kk < 7; kk++)
      acc = wmma_bf16(afr[kk], frag_b_ld(xb, 224, kk * 32, lane), acc);
#pragma unroll
    for (int r = 0; r < 8; r++) {
      float s = acc[r];
      bool mk = s >= thr_r[r];
      float g = (mk ? s : 0.f) + bias_r[r];
      float e = __expf(g * 10.0f - rmx[r]);
      float av = mk ? e * dinv[r] : 0.f;
      aout[(size_t)(r + 8 * hi) * 4096 + mt * 16 + (lane & 15)] = f2bf(av);
    }
  }
}

// ---------------------------------------------------------------------------
// 7) agg = attn @ pi : M=8192, N=784, K=4096.
//    One block (8 waves) per 16-row attn tile. The A tile is streamed through
//    a double-buffered LDS stage with global_load_async_to_lds_b128 (ASYNCcnt)
//    so the DMA of chunk c+1 overlaps the WMMA work on chunk c. Each wave
//    owns up to 7 N-tiles (ct = wave + 8j over 49), giving 7 independent
//    accumulators -> the compiler can clause the 14 B-loads then issue 7
//    back-to-back WMMAs per k-step. pi_t (12.8 MB) is L2-resident.
// ---------------------------------------------------------------------------
__global__ void __launch_bounds__(256)
k_agg(const u16* __restrict__ attn, const u16* __restrict__ pit,
      float* __restrict__ agg) {
  __shared__ __align__(16) u16 Abuf[2][16 * 256];   // double-buffered A chunk
  int lane = threadIdx.x & 31, w = threadIdx.x >> 5, hi = lane >> 4;
  int rt = blockIdx.x;                  // 0..511
  int b = rt >> 8, lbase = (rt & 255) * 16;

  const char* abase = (const char*)(attn + (size_t)(b * 4096 + lbase) * 4096);
  const u16* pbase = pit + (size_t)b * 784 * 4096;
  unsigned ldsA = (unsigned)(size_t)(void*)&Abuf[0][0];   // LDS byte offset

  // stage chunk c (16 rows x 256 k = 8 KB) into buffer (c&1): 2x16B per thread
  auto stage = [&](int c) {
    int t = threadIdx.x;
    unsigned dbase = ldsA + (unsigned)((c & 1) * 8192);
#pragma unroll
    for (int q = 0; q < 2; q++) {
      int u = t + q * 256;              // 0..511 16-byte units
      int row = u >> 5;                 // 0..15
      int off = (u & 31) * 16;          // byte offset within 512B row chunk
      async_cp16(dbase + (unsigned)(row * 512 + off),
                 abase + (size_t)row * 8192 + (size_t)c * 512 + off);
    }
  };

  v8f acc[7] = {{}, {}, {}, {}, {}, {}, {}};
  stage(0);
  for (int c = 0; c < 16; c++) {
    if (c + 1 < 16) {
      stage(c + 1);
      wait_async_le2();                 // chunk c landed (in-order completion)
    } else {
      wait_async_0();
    }
    __syncthreads();                    // all waves' portions visible
    const u16* Ald = &Abuf[c & 1][0];
#pragma unroll
    for (int kk = 0; kk < 8; kk++) {
      v16bf afr = frag_a_ld(Ald, 256, kk * 32, lane);
      v16bf bf[7];
#pragma unroll
      for (int j = 0; j < 7; j++) {
        int ct = w + 8 * j;             // wave-uniform -> EXEC stays full
        if (ct < 49)
          bf[j] = frag_b_ld(pbase + (size_t)ct * 16 * 4096, 4096,
                            c * 256 + kk * 32, lane);
      }
#pragma unroll
      for (int j = 0; j < 7; j++) {
        int ct = w + 8 * j;
        if (ct < 49) acc[j] = wmma_bf16(afr, bf[j], acc[j]);
      }
    }
    __syncthreads();                    // done reading buf before re-stage
  }

#pragma unroll
  for (int j = 0; j < 7; j++) {
    int ct = w + 8 * j;
    if (ct < 49) {
      float* orow = agg + (size_t)(b * 4096 + lbase) * 784 + ct * 16 + (lane & 15);
#pragma unroll
      for (int r = 0; r < 8; r++) orow[(size_t)(r + 8 * hi) * 784] = acc[j][r];
    }
  }
}

// ---------------------------------------------------------------------------
// 8) fold (overlap-add / cnt) + conv1x1(W_w) + residual -> out f32
// ---------------------------------------------------------------------------
__global__ void k_fold(const float* __restrict__ agg, const float* __restrict__ bin,
                       const float* __restrict__ Ww, const float* __restrict__ Wb,
                       float* __restrict__ out) {
  __shared__ float w[64 * 16];
  __shared__ float wb[64];
  for (int i = threadIdx.x; i < 64 * 16; i += blockDim.x) w[i] = Ww[i];
  if (threadIdx.x < 64) wb[threadIdx.x] = Wb[threadIdx.x];
  __syncthreads();
  int pix = blockIdx.x * blockDim.x + threadIdx.x;   // 131072
  int b = pix >> 16, y = (pix >> 8) & 255, x = pix & 255;
  int yp = y + 1, xp = x + 1;                         // crop offset (pl=pt=1)
  int yhi = yp >> 2; if (yhi > 63) yhi = 63;
  int ylo = yp >= 6 ? (yp - 3) >> 2 : 0;
  int xhi = xp >> 2; if (xhi > 63) xhi = 63;
  int xlo = xp >= 6 ? (xp - 3) >> 2 : 0;
  float zi[16];
#pragma unroll
  for (int ci = 0; ci < 16; ci++) zi[ci] = 0.f;
  int cnt = 0;
  for (int lh = ylo; lh <= yhi; lh++) {
    int ky = yp - 4 * lh;
    for (int lw = xlo; lw <= xhi; lw++) {
      int kx = xp - 4 * lw;
      cnt++;
      const float* ap = agg + ((size_t)b * 4096 + lh * 64 + lw) * 784 + ky * 7 + kx;
#pragma unroll
      for (int ci = 0; ci < 16; ci++) zi[ci] += ap[ci * 49];
    }
  }
  float inv = 1.0f / (float)cnt;
#pragma unroll
  for (int ci = 0; ci < 16; ci++) zi[ci] *= inv;
#pragma unroll
  for (int co = 0; co < 64; co++) {
    float s = wb[co];
#pragma unroll
    for (int ci = 0; ci < 16; ci++) s = fmaf(w[co * 16 + ci], zi[ci], s);
    size_t o = ((size_t)(b * 64 + co) * 256 + y) * 256 + x;
    out[o] = bin[o] + s;
  }
}

// ---------------------------------------------------------------------------
extern "C" void kernel_launch(void* const* d_in, const int* in_sizes, int n_in,
                              void* d_out, int out_size, void* d_ws, size_t ws_size,
                              hipStream_t stream) {
  (void)in_sizes; (void)n_in; (void)out_size; (void)ws_size;
  const float* bin  = (const float*)d_in[0];
  const float* g_w  = (const float*)d_in[1];
  const float* g_b  = (const float*)d_in[2];
  const float* th_w = (const float*)d_in[3];
  const float* th_b = (const float*)d_in[4];
  const float* W_w  = (const float*)d_in[5];
  const float* W_b  = (const float*)d_in[6];
  const float* fc1w = (const float*)d_in[7];
  const float* fc1b = (const float*)d_in[8];
  const float* fc2w = (const float*)d_in[9];
  const float* fc2b = (const float*)d_in[10];
  const float* thrw = (const float*)d_in[11];
  const float* thrb = (const float*)d_in[12];
  const float* bw   = (const float*)d_in[13];
  const float* bb   = (const float*)d_in[14];
  float* out = (float*)d_out;

  char* p = (char*)d_ws;
  auto alloc = [&](size_t bytes) {
    void* r = (void*)p;
    p += (bytes + 255) & ~(size_t)255;
    return r;
  };
  u16*   b1   = (u16*)alloc(2ull * 16 * 65536 * 2);      // b1 bf16
  u16*   b2   = (u16*)alloc(2ull * 16 * 65536 * 2);      // b2 bf16
  float* thr  = (float*)alloc(2ull * 4096 * 4);
  float* bias = (float*)alloc(2ull * 4096 * 4);
  u16*   fcw  = (u16*)alloc(2ull * 224 * 800 * 2);       // padded fc weights
  float* fcb  = (float*)alloc(2ull * 224 * 4);
  u16*   wif  = (u16*)alloc(2ull * 4096 * 224 * 2);
  u16*   xif  = (u16*)alloc(2ull * 4096 * 224 * 2);
  u16*   pit  = (u16*)alloc(2ull * 784 * 4096 * 2);      // pi transposed [d][m]
  u16*   attn = (u16*)alloc(2ull * 4096 * 4096 * 2);     // 67 MB (fits L2)
  float* agg  = (float*)alloc(2ull * 4096 * 784 * 4);

  k_conv    <<<512,   256, 0, stream>>>(bin, g_w, g_b, th_w, th_b, b1, b2);
  k_thrbias <<<32,    256, 0, stream>>>(bin, thrw, thrb, bw, bb, thr, bias);
  k_prep    <<<1400,  256, 0, stream>>>(fc1w, fc1b, fc2w, fc2b, fcw, fcb);
  k_patch_pi<<<25088, 256, 0, stream>>>(b2, pit);
  k_fc      <<<512,   224, 0, stream>>>(b1, fcw, fcb, wif, xif);
  k_attn    <<<64,    256, 0, stream>>>(wif, xif, thr, bias, attn);
  k_agg     <<<512,   256, 0, stream>>>(attn, pit, agg);
  k_fold    <<<512,   256, 0, stream>>>(agg, bin, W_w, W_b, out);
}
